// TrajectoryGAT_21655225106673
// MI455X (gfx1250) — matrix operand order repaired and checked by reading, not compiled, not run on hardware
//
#include <hip/hip_runtime.h>
#include <hip/hip_bf16.h>
#include <math.h>

typedef __attribute__((ext_vector_type(16))) _Float16 v16h;
typedef __attribute__((ext_vector_type(8)))  _Float16 v8h;
typedef __attribute__((ext_vector_type(8)))  float    v8f;

#define N_NODES 100000
#define N_EDGES 1600000
#define N_GRAPHS 256
#define HID 128
#define NEG_SLOPE 0.2f

// ---------------------------------------------------------------------------
// WMMA GEMM, f16 in / f32 out. A: MxKp row-major f16 (K zero-padded to Kp,
// Kp % 32 == 0). BT: NxKp row-major f16 (= B transposed, zero-padded).
// C: MxN row-major f32. One wave computes one 16x16 tile; inner loop is
// 4x global_load_b128 + 1x v_wmma_f32_16x16x32_f16, no predication.
// ---------------------------------------------------------------------------
__global__ void wmma_gemm_f16(const _Float16* __restrict__ A,
                              const _Float16* __restrict__ BT,
                              float* __restrict__ C,
                              int M, int N, int Kp) {
    const int wave = (blockIdx.x * blockDim.x + threadIdx.x) >> 5;
    const int lane = threadIdx.x & 31;
    const int tiles_n = N >> 4;
    if (wave >= (M >> 4) * tiles_n) return;      // wave-uniform: EXEC stays full
    const int tm = (wave / tiles_n) << 4;
    const int tn = (wave % tiles_n) << 4;

    const int halfsel = lane >> 4;               // 0: lanes 0-15, 1: lanes 16-31
    const int l16     = lane & 15;
    // 16-bit A 16x32 layout: lane holds row M=l16, K runs {kb..kb+7, kb+16..kb+23},
// kb = 8*halfsel. 16-bit B 32x16 layout: lane holds col N=l16, K runs
// {kb..kb+7, kb+16..kb+23}, kb = 8*halfsel.  (ISA 7.12.2)
    const _Float16* arow = A  + (size_t)(tm + l16) * Kp + halfsel * 8;
    const _Float16* bcol = BT + (size_t)(tn + l16) * Kp + halfsel * 8;

    v8f acc = {};
    for (int kk = 0; kk < Kp; kk += 32) {
        const v8h a_lo = *(const v8h*)(arow + kk);        // K = kb .. kb+7
        const v8h a_hi = *(const v8h*)(arow + kk + 16);   // K = kb+16 .. kb+23
        const v8h b_lo = *(const v8h*)(bcol + kk);        // K = kb .. kb+7
        const v8h b_hi = *(const v8h*)(bcol + kk + 16);   // K = kb+16 .. kb+23
        v16h af, bf;
#pragma unroll
        for (int i = 0; i < 8; ++i) {
            af[i] = a_lo[i]; af[i + 8] = a_hi[i];
            bf[i] = b_lo[i]; bf[i + 8] = b_hi[i];
        }
        acc = __builtin_amdgcn_wmma_f32_16x16x32_f16(
            false, af, false, bf, (short)0, acc, false, false);
    }
    // C/D layout: VGPR d holds M = tm + d + 8*halfsel, N = tn + l16
#pragma unroll
    for (int d = 0; d < 8; ++d)
        C[(size_t)(tm + d + halfsel * 8) * N + tn + l16] = acc[d];
}

// dst[m][k<K] = (f16)src[m][k], zero-padded to stride Kp
__global__ void cvt_pad_f16(const float* __restrict__ src, _Float16* __restrict__ dst,
                            int M, int K, int Kp) {
    int i = blockIdx.x * blockDim.x + threadIdx.x;
    if (i >= M * Kp) return;
    int m = i / Kp, k = i - m * Kp;
    dst[i] = (k < K) ? (_Float16)src[m * K + k] : (_Float16)0.0f;
}

// BT[n][k<K] = (f16)B[k][n], zero-padded to stride Kp
__global__ void cvt_transpose_f16(const float* __restrict__ B, _Float16* __restrict__ BT,
                                  int K, int N, int Kp) {
    int i = blockIdx.x * blockDim.x + threadIdx.x;
    if (i >= N * Kp) return;
    int n = i / Kp, k = i - n * Kp;
    BT[i] = (k < K) ? (_Float16)B[k * N + n] : (_Float16)0.0f;
}

// ---------------------------------------------------------------------------
// Per-node attention scores: a_s[n]=dot(h[n],att_src), a_d[n]=dot(h[n],att_dst)
// ---------------------------------------------------------------------------
__global__ void att_scores(const float* __restrict__ h,
                           const float* __restrict__ att_s,
                           const float* __restrict__ att_d,
                           float* __restrict__ a_s, float* __restrict__ a_d,
                           int n_nodes) {
    const int wave = (blockIdx.x * blockDim.x + threadIdx.x) >> 5;
    const int lane = threadIdx.x & 31;
    if (wave >= n_nodes) return;
    const float4 hv = ((const float4*)(h + (size_t)wave * HID))[lane];
    const float4 sv = ((const float4*)att_s)[lane];
    const float4 dv = ((const float4*)att_d)[lane];
    float ss = hv.x * sv.x + hv.y * sv.y + hv.z * sv.z + hv.w * sv.w;
    float dd = hv.x * dv.x + hv.y * dv.y + hv.z * dv.z + hv.w * dv.w;
#pragma unroll
    for (int off = 16; off > 0; off >>= 1) {
        ss += __shfl_xor(ss, off, 32);
        dd += __shfl_xor(dd, off, 32);
    }
    if (lane == 0) { a_s[wave] = ss; a_d[wave] = dd; }
}

__global__ void fill_f32(float* __restrict__ p, float v, int n) {
    int i = blockIdx.x * blockDim.x + threadIdx.x;
    if (i < n) p[i] = v;
}

// Race-safe float atomic-max via int-max (positive) / uint-min (negative).
__device__ __forceinline__ void atomicMaxF(float* addr, float val) {
    if (val >= 0.0f)
        atomicMax((int*)addr, __float_as_int(val));
    else
        atomicMin((unsigned int*)addr, (unsigned int)__float_as_int(val));
}

// alpha_e = leaky_relu(a_s[src] + a_d[dst]); m[dst] = max(m[dst], alpha)
__global__ void edge_alpha_max(const int* __restrict__ src, const int* __restrict__ dst,
                               const float* __restrict__ a_s, const float* __restrict__ a_d,
                               float* __restrict__ alpha, float* __restrict__ m, int E) {
    int e = blockIdx.x * blockDim.x + threadIdx.x;
    if (e >= E) return;
    float a = a_s[src[e]] + a_d[dst[e]];
    a = (a > 0.0f) ? a : NEG_SLOPE * a;
    alpha[e] = a;
    atomicMaxF(&m[dst[e]], a);
}

// e = exp(alpha - m[dst]); denom[dst] += e   (alpha overwritten in place)
__global__ void edge_expsum(const int* __restrict__ dst, float* __restrict__ alpha,
                            const float* __restrict__ m, float* __restrict__ denom, int E) {
    int e = blockIdx.x * blockDim.x + threadIdx.x;
    if (e >= E) return;
    int d = dst[e];
    float ev = __expf(alpha[e] - m[d]);
    alpha[e] = ev;
    atomicAdd(&denom[d], ev);
}

// out[dst] += h[src] * (e/(denom[dst]+eps)); one wave per edge, float4/lane.
__global__ void edge_scatter(const int* __restrict__ src, const int* __restrict__ dst,
                             const float* __restrict__ expv, const float* __restrict__ denom,
                             const float* __restrict__ h, float* __restrict__ out, int E) {
    const int wave = (blockIdx.x * blockDim.x + threadIdx.x) >> 5;
    const int lane = threadIdx.x & 31;
    if (wave >= E) return;
    const int s = src[wave], d = dst[wave];
    const float w = expv[wave] / (denom[d] + 1e-16f);
    const float4 hv = ((const float4*)(h + (size_t)s * HID))[lane];
    float* o = out + (size_t)d * HID + lane * 4;
    atomicAdd(o + 0, hv.x * w);
    atomicAdd(o + 1, hv.y * w);
    atomicAdd(o + 2, hv.z * w);
    atomicAdd(o + 3, hv.w * w);
}

// layer-1 epilogue: v = relu(acc + bias); write f16 GEMM input for layer 2
__global__ void bias_relu_cvt_f16(const float* __restrict__ acc, const float* __restrict__ bias,
                                  _Float16* __restrict__ dst, int n_elems) {
    int i = blockIdx.x * blockDim.x + threadIdx.x;
    if (i >= n_elems) return;
    dst[i] = (_Float16)fmaxf(acc[i] + bias[i & (HID - 1)], 0.0f);
}

// layer-2 epilogue: out += bias (f32, in place)
__global__ void bias_add(float* __restrict__ out, const float* __restrict__ bias, int n_elems) {
    int i = blockIdx.x * blockDim.x + threadIdx.x;
    if (i >= n_elems) return;
    out[i] += bias[i & (HID - 1)];
}

// sums[batch[n]] += h[n]; cnt[batch[n]] += 1. One wave per node.
__global__ void pool_kernel(const float* __restrict__ h, const int* __restrict__ batch,
                            float* __restrict__ sums, float* __restrict__ cnt, int n_nodes) {
    const int wave = (blockIdx.x * blockDim.x + threadIdx.x) >> 5;
    const int lane = threadIdx.x & 31;
    if (wave >= n_nodes) return;
    const int g = batch[wave];
    const float4 hv = ((const float4*)(h + (size_t)wave * HID))[lane];
    float* s = sums + (size_t)g * HID + lane * 4;
    atomicAdd(s + 0, hv.x);
    atomicAdd(s + 1, hv.y);
    atomicAdd(s + 2, hv.z);
    atomicAdd(s + 3, hv.w);
    if (lane == 0) atomicAdd(&cnt[g], 1.0f);
}

// pooled = sums/max(cnt,1); logits = pooled@Wc + bc; log_softmax. Wave/graph.
__global__ void head_kernel(const float* __restrict__ sums, const float* __restrict__ cnt,
                            const float* __restrict__ Wc, const float* __restrict__ bc,
                            float* __restrict__ out, int n_graphs) {
    const int wave = (blockIdx.x * blockDim.x + threadIdx.x) >> 5;
    const int lane = threadIdx.x & 31;
    if (wave >= n_graphs) return;
    const float4 p = ((const float4*)(sums + (size_t)wave * HID))[lane];
    const int k = lane * 4;
    float l0 = p.x * Wc[(k + 0) * 2 + 0] + p.y * Wc[(k + 1) * 2 + 0] +
               p.z * Wc[(k + 2) * 2 + 0] + p.w * Wc[(k + 3) * 2 + 0];
    float l1 = p.x * Wc[(k + 0) * 2 + 1] + p.y * Wc[(k + 1) * 2 + 1] +
               p.z * Wc[(k + 2) * 2 + 1] + p.w * Wc[(k + 3) * 2 + 1];
#pragma unroll
    for (int off = 16; off > 0; off >>= 1) {
        l0 += __shfl_xor(l0, off, 32);
        l1 += __shfl_xor(l1, off, 32);
    }
    if (lane == 0) {
        const float c = fmaxf(cnt[wave], 1.0f);
        l0 = l0 / c + bc[0];
        l1 = l1 / c + bc[1];
        const float mx  = fmaxf(l0, l1);
        const float lse = mx + logf(__expf(l0 - mx) + __expf(l1 - mx));
        out[wave * 2 + 0] = l0 - lse;
        out[wave * 2 + 1] = l1 - lse;
    }
}

// ---------------------------------------------------------------------------
static inline int cdiv(long long a, long long b) { return (int)((a + b - 1) / b); }

extern "C" void kernel_launch(void* const* d_in, const int* in_sizes, int n_in,
                              void* d_out, int out_size, void* d_ws, size_t ws_size,
                              hipStream_t stream) {
    const float* x      = (const float*)d_in[0];
    const int*   eidxS  = (const int*)  d_in[1];
    const int*   eidxT  = (const int*)  d_in[3];
    const int*   batch  = (const int*)  d_in[5];
    const float* W1     = (const float*)d_in[6];
    const float* as1    = (const float*)d_in[7];
    const float* ad1    = (const float*)d_in[8];
    const float* b1     = (const float*)d_in[9];
    const float* W2     = (const float*)d_in[10];
    const float* as2    = (const float*)d_in[11];
    const float* ad2    = (const float*)d_in[12];
    const float* b2     = (const float*)d_in[13];
    const float* Wc     = (const float*)d_in[14];
    const float* bc     = (const float*)d_in[15];
    float* out = (float*)d_out;

    const long long NH = (long long)N_NODES * HID;   // 12,800,000

    // Workspace layout
    float*    bufA  = (float*)d_ws;                     // h (GEMM out)      [NH]
    float*    bufB  = bufA + NH;                        // accumulators      [NH]
    float*    a_s   = bufB + NH;                        // [N_NODES]
    float*    a_d   = a_s + N_NODES;
    float*    mArr  = a_d + N_NODES;
    float*    denom = mArr + N_NODES;
    float*    alpha = denom + N_NODES;                  // [N_EDGES]
    float*    sums  = alpha + N_EDGES;                  // [N_GRAPHS*HID]
    float*    cnt   = sums + (size_t)N_GRAPHS * HID;    // [N_GRAPHS]
    _Float16* xA16  = (_Float16*)(cnt + N_GRAPHS);      // x padded  [N_NODES*32]
    _Float16* hA16  = xA16 + (size_t)N_NODES * 32;      // layer2 in [NH]
    _Float16* w1T   = hA16 + NH;                        // [HID*32]
    _Float16* w2T   = w1T + HID * 32;                   // [HID*HID]

    const int TB = 256;
    const int gemm_blocks      = cdiv((N_NODES / 16) * (HID / 16), TB / 32); // 6,250
    const int node_wave_blocks = cdiv(N_NODES, TB / 32);                     // 12,500
    const int edge_blocks      = cdiv(N_EDGES, TB);                          // 6,250
    const int edge_wave_blocks = cdiv(N_EDGES, TB / 32);                     // 200,000
    const int nh_blocks        = cdiv(NH, TB);                               // 50,000
    const int node_blocks      = cdiv(N_NODES, TB);

    // ===== Layer 1 (spatial): h1 = x @ W1 =====
    cvt_pad_f16<<<cdiv((long long)N_NODES * 32, TB), TB, 0, stream>>>(x, xA16, N_NODES, 9, 32);
    cvt_transpose_f16<<<cdiv(HID * 32, TB), TB, 0, stream>>>(W1, w1T, 9, HID, 32);
    wmma_gemm_f16<<<gemm_blocks, TB, 0, stream>>>(xA16, w1T, bufA, N_NODES, HID, 32);
    att_scores<<<node_wave_blocks, TB, 0, stream>>>(bufA, as1, ad1, a_s, a_d, N_NODES);
    fill_f32<<<node_blocks, TB, 0, stream>>>(mArr, -INFINITY, N_NODES);
    fill_f32<<<node_blocks, TB, 0, stream>>>(denom, 0.0f, N_NODES);
    fill_f32<<<nh_blocks, TB, 0, stream>>>(bufB, 0.0f, (int)NH);
    edge_alpha_max<<<edge_blocks, TB, 0, stream>>>(eidxS, eidxS + N_EDGES,
                                                   a_s, a_d, alpha, mArr, N_EDGES);
    edge_expsum<<<edge_blocks, TB, 0, stream>>>(eidxS + N_EDGES, alpha, mArr, denom, N_EDGES);
    edge_scatter<<<edge_wave_blocks, TB, 0, stream>>>(eidxS, eidxS + N_EDGES,
                                                      alpha, denom, bufA, bufB, N_EDGES);
    bias_relu_cvt_f16<<<nh_blocks, TB, 0, stream>>>(bufB, b1, hA16, (int)NH);

    // ===== Layer 2 (temporal): h2 = relu_h @ W2 =====
    cvt_transpose_f16<<<cdiv(HID * HID, TB), TB, 0, stream>>>(W2, w2T, HID, HID, HID);
    wmma_gemm_f16<<<gemm_blocks, TB, 0, stream>>>(hA16, w2T, bufA, N_NODES, HID, HID);
    att_scores<<<node_wave_blocks, TB, 0, stream>>>(bufA, as2, ad2, a_s, a_d, N_NODES);
    fill_f32<<<node_blocks, TB, 0, stream>>>(mArr, -INFINITY, N_NODES);
    fill_f32<<<node_blocks, TB, 0, stream>>>(denom, 0.0f, N_NODES);
    fill_f32<<<nh_blocks, TB, 0, stream>>>(bufB, 0.0f, (int)NH);
    edge_alpha_max<<<edge_blocks, TB, 0, stream>>>(eidxT, eidxT + N_EDGES,
                                                   a_s, a_d, alpha, mArr, N_EDGES);
    edge_expsum<<<edge_blocks, TB, 0, stream>>>(eidxT + N_EDGES, alpha, mArr, denom, N_EDGES);
    edge_scatter<<<edge_wave_blocks, TB, 0, stream>>>(eidxT, eidxT + N_EDGES,
                                                      alpha, denom, bufA, bufB, N_EDGES);
    bias_add<<<nh_blocks, TB, 0, stream>>>(bufB, b2, (int)NH);

    // ===== Pool + classifier head =====
    fill_f32<<<cdiv(N_GRAPHS * HID, TB), TB, 0, stream>>>(sums, 0.0f, N_GRAPHS * HID);
    fill_f32<<<1, TB, 0, stream>>>(cnt, 0.0f, N_GRAPHS);
    pool_kernel<<<node_wave_blocks, TB, 0, stream>>>(bufB, batch, sums, cnt, N_NODES);
    head_kernel<<<cdiv(N_GRAPHS, TB / 32), TB, 0, stream>>>(sums, cnt, Wc, bc, out, N_GRAPHS);
}
